// eVONet_46196668235892
// MI455X (gfx1250) — matrix operand order repaired and checked by reading, not compiled, not run on hardware
//
#include <hip/hip_runtime.h>
#include <hip/hip_bf16.h>

// ---------------------------------------------------------------------------
// eVONet update operator for MI455X (gfx1250, wave32, WMMA, async-to-LDS).
// GEMMs: bf16 inputs, f32 accumulation via v_wmma_f32_16x16x32_bf16.
// Weight/A tiles staged with GLOBAL_LOAD_ASYNC_TO_LDS_B128, double-buffered.
// All reductions / layernorm / softmax epilogues in f32.
// ---------------------------------------------------------------------------

#define E       98304
#define DIMN    384
#define NPATCH  7680
#define NFRAME  96
#define NIJ     (NFRAME * NFRAME)   // 9216
#define KCORR   882
#define KCORRP  896                  // padded to multiple of 32

typedef __bf16 bf16;
typedef __attribute__((ext_vector_type(16))) __bf16 bf16x16;
typedef __attribute__((ext_vector_type(8)))  __bf16 bf16x8;
typedef __attribute__((ext_vector_type(8)))  float  f32x8;

// ---- CDNA5 async copy helpers (ASYNCcnt path) -----------------------------
__device__ __forceinline__ unsigned lds_off(const void* p) {
    // generic LDS address: addr[31:0] is the LDS byte offset (flat aperture)
    return (unsigned)(unsigned long long)p;
}
__device__ __forceinline__ void async_b128(unsigned loff, unsigned long long ga) {
    asm volatile("global_load_async_to_lds_b128 %0, %1, off"
                 :: "v"(loff), "v"(ga) : "memory");
}
__device__ __forceinline__ void wait_async0() {
    asm volatile("s_wait_asynccnt 0" ::: "memory");
}

__device__ __forceinline__ float fast_sigmoid(float v) {
    return __builtin_amdgcn_rcpf(1.f + __expf(-v));
}

// ---------------------------------------------------------------------------
// Tiled GEMM: out[M x 384] = act(A[M x K] @ W[K x 384] + bias)
// WT is the weight stored TRANSPOSED in global: WT[n*K + k] = W[k*384 + n].
// block = 256 threads (8 waves). M-block = 32 rows. Each wave: 16 rows x 96
// cols = 6 WMMA accumulators. W chunk (384x32) and A chunk (32x32) staged
// into double-buffered LDS via global_load_async_to_lds_b128.
// M and K must be multiples of 32. act: 0=none 1=relu 2=sigmoid
// ---------------------------------------------------------------------------
__global__ __launch_bounds__(256) void k_gemm(const bf16* __restrict__ A,
                                              const bf16* __restrict__ WT,
                                              const float* __restrict__ bias,
                                              float* __restrict__ out,
                                              int K, int act)
{
    __shared__ bf16 WtS[2][DIMN * 32];   // [n][klocal], klocal contiguous
    __shared__ bf16 AbS[2][32 * 32];     // [rowlocal][klocal]

    const int tid  = threadIdx.x;
    const int lane = tid & 31;
    const int wave = tid >> 5;
    const int wm   = wave & 1;           // 2 M-tiles
    const int wn   = wave >> 1;          // 4 col groups of 96

    const int rblk  = blockIdx.x * 32;
    const int cbase = wn * 96;

    // A fragment addressing (16-bit A 16x32 layout, wave32)
    const int arowl = wm * 16 + (lane & 15);
    const int kb    = (lane >> 4) * 16;
    // B fragment addressing (16-bit B 32x16 layout, wave32)
    const int bcol  = lane & 15;
    const int ko    = (lane >> 4) * 16;

    // async staging of chunk kc into buffer buf
    auto prefetch = [&](int kc, int buf) {
        const long kofs = (long)kc * 32;
#pragma unroll
        for (int i = 0; i < 6; ++i) {              // W: 1536 b128 / 256 thr
            const int idx = tid + i * 256;
            const int n = idx >> 2, seg = idx & 3;
            async_b128(lds_off(&WtS[buf][n * 32 + seg * 8]),
                       (unsigned long long)(WT + (long)n * K + kofs + seg * 8));
        }
        if (tid < 128) {                            // A: 128 b128
            const int row = tid >> 2, seg = tid & 3;
            async_b128(lds_off(&AbS[buf][row * 32 + seg * 8]),
                       (unsigned long long)(A + (long)(rblk + row) * K + kofs + seg * 8));
        }
    };

    f32x8 acc[6];
#pragma unroll
    for (int t = 0; t < 6; ++t)
#pragma unroll
        for (int r = 0; r < 8; ++r) acc[t][r] = 0.f;

    const int nk = K >> 5;
    prefetch(0, 0);
    for (int kc = 0; kc < nk; ++kc) {
        const int cur = kc & 1;
        wait_async0();          // this wave's async portions have landed
        __syncthreads();        // all waves' portions landed; prev reads done
        if (kc + 1 < nk) prefetch(kc + 1, cur ^ 1);

        // A fragment from LDS
        const bf16* ap = &AbS[cur][arowl * 32];
        bf16x8 alo = *(const bf16x8*)(ap + kb);
        bf16x8 ahi = *(const bf16x8*)(ap + kb + 8);
        bf16x16 a;
#pragma unroll
        for (int i = 0; i < 8; ++i) { a[i] = alo[i]; a[8 + i] = ahi[i]; }

#pragma unroll
        for (int t = 0; t < 6; ++t) {
            const bf16* bp = &WtS[cur][(cbase + t * 16 + bcol) * 32 + ko];
            bf16x8 blo = *(const bf16x8*)bp;
            bf16x8 bhi = *(const bf16x8*)(bp + 8);
            bf16x16 b;
#pragma unroll
            for (int i = 0; i < 8; ++i) { b[i] = blo[i]; b[8 + i] = bhi[i]; }
            acc[t] = __builtin_amdgcn_wmma_f32_16x16x32_bf16(
                false, a, false, b, (short)0, acc[t], false, false);
        }
    }

    // epilogue: C layout -> row = rblk + wm*16 + (lane>>4)*8 + r
    const int orow = rblk + wm * 16 + (lane >> 4) * 8;
    const int ocol = cbase + (lane & 15);
#pragma unroll
    for (int t = 0; t < 6; ++t) {
        const int cc = ocol + t * 16;
        const float bb = bias[cc];
#pragma unroll
        for (int r = 0; r < 8; ++r) {
            float v = acc[t][r] + bb;
            if (act == 1)      v = fmaxf(v, 0.f);
            else if (act == 2) v = fast_sigmoid(v);
            out[(long)(orow + r) * DIMN + cc] = v;
        }
    }
}

// ---------------------------------------------------------------------------
// LayerNorm over rows of 384 (eps = 1e-3). One wave per row, 12 f32/lane.
// Optional pre-adds (x = in + add0 + add1), optional post-relu.
// Writes f32 and/or bf16 outputs (nullable).
// ---------------------------------------------------------------------------
__global__ __launch_bounds__(256) void k_ln(const float* __restrict__ in,
                                            const float* __restrict__ add0,
                                            const float* __restrict__ add1,
                                            const float* __restrict__ g,
                                            const float* __restrict__ b,
                                            float* __restrict__ outf,
                                            bf16* __restrict__ outb,
                                            int postrelu)
{
    const int lane = threadIdx.x & 31;
    const int wave = threadIdx.x >> 5;
    const long row = (long)blockIdx.x * 8 + wave;
    const long base = row * DIMN;

    float v[12];
    float s = 0.f;
#pragma unroll
    for (int i = 0; i < 12; ++i) {
        const int c = lane + i * 32;
        float x = in[base + c];
        if (add0) x += add0[base + c];
        if (add1) x += add1[base + c];
        v[i] = x;
        s += x;
    }
#pragma unroll
    for (int o = 16; o > 0; o >>= 1) s += __shfl_xor(s, o, 32);
    const float m = s * (1.f / 384.f);
    float vs = 0.f;
#pragma unroll
    for (int i = 0; i < 12; ++i) { const float d = v[i] - m; vs += d * d; }
#pragma unroll
    for (int o = 16; o > 0; o >>= 1) vs += __shfl_xor(vs, o, 32);
    const float inv = rsqrtf(vs * (1.f / 384.f) + 1e-3f);

#pragma unroll
    for (int i = 0; i < 12; ++i) {
        const int c = lane + i * 32;
        float y = (v[i] - m) * inv * g[c] + b[c];
        if (postrelu) y = fmaxf(y, 0.f);
        if (outf) outf[base + c] = y;
        if (outb) outb[base + c] = (bf16)y;
    }
}

// ------------------------- elementwise helpers -----------------------------
__global__ void k_f2bf(const float* __restrict__ in, bf16* __restrict__ out, long n)
{
    long i = (long)blockIdx.x * blockDim.x + threadIdx.x;
    if (i < n) out[i] = (bf16)in[i];
}

// weight [K x 384] -> bf16 transposed [384 x K]
__global__ void k_w2bf_t(const float* __restrict__ w, bf16* __restrict__ out)
{
    long i = (long)blockIdx.x * blockDim.x + threadIdx.x;
    if (i >= (long)DIMN * DIMN) return;
    const int n = (int)(i / DIMN);
    const int k = (int)(i % DIMN);
    out[i] = (bf16)w[(long)k * DIMN + n];
}

// corr_w1 [882 x 384] -> bf16 transposed+padded [384 x 896]
__global__ void k_wpad_t(const float* __restrict__ w, bf16* __restrict__ out)
{
    long i = (long)blockIdx.x * blockDim.x + threadIdx.x;
    if (i >= (long)DIMN * KCORRP) return;
    const int n = (int)(i / KCORRP);
    const int k = (int)(i % KCORRP);
    out[i] = (bf16)(k < KCORR ? w[(long)k * DIMN + n] : 0.f);
}

__global__ void k_corrpad(const float* __restrict__ corr, bf16* __restrict__ out)
{
    long i = (long)blockIdx.x * blockDim.x + threadIdx.x;
    if (i >= (long)E * KCORRP) return;
    const long e = i / KCORRP;
    const int  k = (int)(i % KCORRP);
    out[i] = (bf16)(k < KCORR ? corr[e * KCORR + k] : 0.f);
}

__global__ void k_gather_bf(const float* __restrict__ X, const int* __restrict__ idx,
                            bf16* __restrict__ out, long n)
{
    long i = (long)blockIdx.x * blockDim.x + threadIdx.x;
    if (i >= n) return;
    const long e = i / DIMN;
    const int  c = (int)(i % DIMN);
    const int  s = idx[e];
    out[i] = (bf16)(s >= 0 ? X[(long)s * DIMN + c] : 0.f);
}

__global__ void k_add(float* __restrict__ X, const float* __restrict__ T, long n)
{
    long i = (long)blockIdx.x * blockDim.x + threadIdx.x;
    if (i < n) X[i] += T[i];
}

__global__ void k_gres(float* __restrict__ X, const float* __restrict__ G,
                       const float* __restrict__ R, long n)
{
    long i = (long)blockIdx.x * blockDim.x + threadIdx.x;
    if (i < n) X[i] += G[i] * R[i];
}

__global__ void k_scatadd(float* __restrict__ X, const float* __restrict__ H,
                          const int* __restrict__ seg, long n)
{
    long i = (long)blockIdx.x * blockDim.x + threadIdx.x;
    if (i >= n) return;
    const long e = i / DIMN;
    const int  c = (int)(i % DIMN);
    X[i] += H[(long)seg[e] * DIMN + c];
}

__global__ void k_mkseg(const int* __restrict__ ii, const int* __restrict__ jj,
                        int* __restrict__ seg)
{
    long i = (long)blockIdx.x * blockDim.x + threadIdx.x;
    if (i < E) seg[i] = ii[i] * NFRAME + jj[i];
}

// ------------------------- segment softmax ---------------------------------
__device__ inline unsigned ord_enc(float f) {
    unsigned u = __float_as_uint(f);
    return (u & 0x80000000u) ? ~u : (u | 0x80000000u);
}
__device__ inline float ord_dec(unsigned m) {
    return __uint_as_float((m & 0x80000000u) ? (m ^ 0x80000000u) : ~m);
}

__global__ void k_seginit(unsigned* __restrict__ mx, float* __restrict__ den,
                          float* __restrict__ y, long n)
{
    long i = (long)blockIdx.x * blockDim.x + threadIdx.x;
    if (i >= n) return;
    mx[i]  = 0x007FFFFFu;   // ord_enc(-inf)
    den[i] = 0.f;
    y[i]   = 0.f;
}

__global__ void k_segmax(const float* __restrict__ g, const int* __restrict__ seg,
                         unsigned* __restrict__ mx, long n)
{
    long i = (long)blockIdx.x * blockDim.x + threadIdx.x;
    if (i >= n) return;
    const long e = i / DIMN;
    const int  c = (int)(i % DIMN);
    atomicMax(&mx[(long)seg[e] * DIMN + c], ord_enc(g[i]));
}

__global__ void k_segexp(float* __restrict__ g, const int* __restrict__ seg,
                         const unsigned* __restrict__ mx, float* __restrict__ den,
                         long n)
{
    long i = (long)blockIdx.x * blockDim.x + threadIdx.x;
    if (i >= n) return;
    const long e = i / DIMN;
    const int  c = (int)(i % DIMN);
    const long s = (long)seg[e] * DIMN + c;
    const float ex = __expf(g[i] - ord_dec(mx[s]));
    g[i] = ex;
    atomicAdd(&den[s], ex);
}

__global__ void k_segwsum(const float* __restrict__ f, const float* __restrict__ g,
                          const int* __restrict__ seg, const float* __restrict__ den,
                          float* __restrict__ y, long n)
{
    long i = (long)blockIdx.x * blockDim.x + threadIdx.x;
    if (i >= n) return;
    const long e = i / DIMN;
    const int  c = (int)(i % DIMN);
    const long s = (long)seg[e] * DIMN + c;
    atomicAdd(&y[s], f[i] * g[i] * __builtin_amdgcn_rcpf(den[s]));
}

// ------------------------- output heads (N=2, VALU) ------------------------
__global__ void k_heads(const float* __restrict__ X,
                        const float* __restrict__ dw, const float* __restrict__ db,
                        const float* __restrict__ ww, const float* __restrict__ wb,
                        float* __restrict__ dout, float* __restrict__ wout)
{
    long i = (long)blockIdx.x * blockDim.x + threadIdx.x;
    if (i >= (long)E * 4) return;
    const long e = i >> 2;
    const int  j = (int)(i & 3);
    const int  o = j & 1;
    const float* xr = X + e * DIMN;
    const float* wm = (j < 2) ? dw : ww;
    float s = (j < 2) ? db[o] : wb[o];
    for (int k = 0; k < DIMN; ++k) s += fmaxf(xr[k], 0.f) * wm[k * 2 + o];
    if (j < 2) dout[e * 2 + o] = s;
    else       wout[e * 2 + o] = fast_sigmoid(s);
}

// ---------------------------------------------------------------------------
extern "C" void kernel_launch(void* const* d_in, const int* in_sizes, int n_in,
                              void* d_out, int out_size, void* d_ws, size_t ws_size,
                              hipStream_t stream)
{
    (void)in_sizes; (void)n_in; (void)out_size; (void)ws_size;

    // ---- inputs (setup_inputs dict order; params flattened in dict order) --
    const float* net  = (const float*)d_in[0];
    const float* inp  = (const float*)d_in[1];
    const float* corr = (const float*)d_in[2];
    const int*   ii   = (const int*)d_in[4];
    const int*   jj   = (const int*)d_in[5];
    const int*   kk   = (const int*)d_in[6];
    const int*   ix   = (const int*)d_in[7];
    const int*   jx   = (const int*)d_in[8];
    const float** P = (const float**)(d_in + 9);   // params start here
    //  0 corr_w1  1 corr_b1  2 corr_w2  3 corr_b2  4 corr_ln_g 5 corr_ln_b
    //  6 corr_w3  7 corr_b3  8 norm_g   9 norm_b
    // 10 c1_w1 11 c1_b1 12 c1_w2 13 c1_b2 14 c2_w1 15 c2_b1 16 c2_w2 17 c2_b2
    // 18 kkf_w 19 kkf_b 20 kkg_w 21 kkg_b 22 kkh_w 23 kkh_b
    // 24 ijf_w 25 ijf_b 26 ijg_w 27 ijg_b 28 ijh_w 29 ijh_b
    // 30 ln1_g 31 ln1_b 32 g1gate_w 33 g1gate_b 34 g1r1_w 35 g1r1_b 36 g1r2_w 37 g1r2_b
    // 38 ln2_g 39 ln2_b 40 g2gate_w 41 g2gate_b 42 g2r1_w 43 g2r1_b 44 g2r2_w 45 g2r2_b
    // 46 d_w 47 d_b 48 w_w 49 w_b

    // ---- workspace layout --------------------------------------------------
    const long Ef = (long)E * DIMN;
    float* T0  = (float*)d_ws;
    float* T1  = T0 + Ef;
    bf16*  ABF = (bf16*)(T1 + Ef);                 // E x 896 bf16 activations
    bf16*  WBF = ABF + (long)E * KCORRP;           // bf16 weights (transposed)
    unsigned* SEGMX = (unsigned*)(WBF + (long)(KCORRP + 18 * DIMN) * DIMN);
    float* SEGDEN = (float*)(SEGMX + (long)NIJ * DIMN);
    float* SEGY   = SEGDEN + (long)NIJ * DIMN;
    float* SEGH   = SEGY + (long)NIJ * DIMN;
    int*   SEGIJ  = (int*)(SEGH + (long)NIJ * DIMN);

    // x lives directly in d_out (fully overwritten before first read)
    float* X    = (float*)d_out;
    float* Dout = X + Ef;
    float* Wout = Dout + (long)E * 2;

    const int TB = 256;
    auto grid = [](long n) { return (unsigned)((n + 255) / 256); };
    const unsigned gE   = (unsigned)(E / 32);       // 3072 GEMM row-blocks
    const unsigned gKK  = (unsigned)(NPATCH / 32);  // 240
    const unsigned gIJ  = (unsigned)(NIJ / 32);     // 288
    const unsigned gLN  = (unsigned)(E / 8);        // 12288

    // bf16 weight slots (transposed): w1t at 0, then 18 DIMxDIM matrices
    bf16* W1T = WBF;
    auto WB = [&](int k) { return WBF + (long)KCORRP * DIMN + (long)k * DIMN * DIMN; };

    // ---- convert weights to bf16 (transposed) ------------------------------
    k_wpad_t<<<grid((long)DIMN * KCORRP), TB, 0, stream>>>(P[0], W1T);
    const int wsrc[18] = { 2, 6, 10, 12, 14, 16, 18, 20, 22, 24, 26, 28,
                           32, 34, 36, 40, 42, 44 };
    for (int k = 0; k < 18; ++k)
        k_w2bf_t<<<grid((long)DIMN * DIMN), TB, 0, stream>>>(P[wsrc[k]], WB(k));

    // ---- corr encoder ------------------------------------------------------
    k_corrpad<<<grid((long)E * KCORRP), TB, 0, stream>>>(corr, ABF);
    k_gemm<<<gE, TB, 0, stream>>>(ABF, W1T, P[1], T0, KCORRP, 1);       // relu
    k_f2bf<<<grid(Ef), TB, 0, stream>>>(T0, ABF, Ef);
    k_gemm<<<gE, TB, 0, stream>>>(ABF, WB(0), P[3], T1, DIMN, 0);
    k_ln<<<gLN, TB, 0, stream>>>(T1, nullptr, nullptr, P[4], P[5],
                                 nullptr, ABF, 1);                      // LN->relu->bf16
    k_gemm<<<gE, TB, 0, stream>>>(ABF, WB(1), P[7], T0, DIMN, 0);
    // x = LN(net + inp + c)
    k_ln<<<gLN, TB, 0, stream>>>(T0, net, inp, P[8], P[9], X, nullptr, 0);

    // ---- neighbor message passing (ix, then jx) ---------------------------
    const int* nidx[2] = { ix, jx };
    const int  c_w1[2] = { 2, 4 }, c_b1[2] = { 11, 15 };
    const int  c_w2[2] = { 3, 5 }, c_b2[2] = { 13, 17 };
    for (int m = 0; m < 2; ++m) {
        k_gather_bf<<<grid(Ef), TB, 0, stream>>>(X, nidx[m], ABF, Ef);
        k_gemm<<<gE, TB, 0, stream>>>(ABF, WB(c_w1[m]), P[c_b1[m]], T0, DIMN, 1);
        k_f2bf<<<grid(Ef), TB, 0, stream>>>(T0, ABF, Ef);
        k_gemm<<<gE, TB, 0, stream>>>(ABF, WB(c_w2[m]), P[c_b2[m]], T1, DIMN, 0);
        k_add<<<grid(Ef), TB, 0, stream>>>(X, T1, Ef);
    }

    // ---- SoftAgg (kk over NPATCH, then ii*96+jj over NIJ) ------------------
    k_mkseg<<<grid(E), TB, 0, stream>>>(ii, jj, SEGIJ);
    const int* segp[2]  = { kk, SEGIJ };
    const long nseg[2]  = { NPATCH, NIJ };
    const unsigned gseg[2] = { gKK, gIJ };
    const int a_fw[2] = { 6, 9 }, a_fb[2] = { 19, 25 };
    const int a_gw[2] = { 7, 10 }, a_gb[2] = { 21, 27 };
    const int a_hw[2] = { 8, 11 }, a_hb[2] = { 23, 29 };
    for (int m = 0; m < 2; ++m) {
        const long ns = nseg[m] * DIMN;
        k_f2bf<<<grid(Ef), TB, 0, stream>>>(X, ABF, Ef);                // xbf
        k_gemm<<<gE, TB, 0, stream>>>(ABF, WB(a_fw[m]), P[a_fb[m]], T0, DIMN, 0); // f
        k_gemm<<<gE, TB, 0, stream>>>(ABF, WB(a_gw[m]), P[a_gb[m]], T1, DIMN, 0); // g
        k_seginit<<<grid(ns), TB, 0, stream>>>(SEGMX, SEGDEN, SEGY, ns);
        k_segmax<<<grid(Ef), TB, 0, stream>>>(T1, segp[m], SEGMX, Ef);
        k_segexp<<<grid(Ef), TB, 0, stream>>>(T1, segp[m], SEGMX, SEGDEN, Ef);
        k_segwsum<<<grid(Ef), TB, 0, stream>>>(T0, T1, segp[m], SEGDEN, SEGY, Ef);
        k_f2bf<<<grid(ns), TB, 0, stream>>>(SEGY, ABF, ns);
        k_gemm<<<gseg[m], TB, 0, stream>>>(ABF, WB(a_hw[m]), P[a_hb[m]], SEGH, DIMN, 0);
        k_scatadd<<<grid(Ef), TB, 0, stream>>>(X, SEGH, segp[m], Ef);
    }

    // ---- gru: LN -> GatedResidual -> LN -> GatedResidual -------------------
    const int ln_g[2] = { 30, 38 }, ln_b[2] = { 31, 39 };
    const int g_gw[2] = { 12, 15 }, g_gb[2] = { 33, 41 };
    const int g_r1w[2] = { 13, 16 }, g_r1b[2] = { 35, 43 };
    const int g_r2w[2] = { 14, 17 }, g_r2b[2] = { 37, 45 };
    for (int m = 0; m < 2; ++m) {
        k_ln<<<gLN, TB, 0, stream>>>(X, nullptr, nullptr, P[ln_g[m]], P[ln_b[m]],
                                     X, ABF, 0);                        // in-place + bf16
        k_gemm<<<gE, TB, 0, stream>>>(ABF, WB(g_gw[m]), P[g_gb[m]], T1, DIMN, 2);  // gate
        k_gemm<<<gE, TB, 0, stream>>>(ABF, WB(g_r1w[m]), P[g_r1b[m]], T0, DIMN, 1);// r1
        k_f2bf<<<grid(Ef), TB, 0, stream>>>(T0, ABF, Ef);
        k_gemm<<<gE, TB, 0, stream>>>(ABF, WB(g_r2w[m]), P[g_r2b[m]], T0, DIMN, 0);// r2
        k_gres<<<grid(Ef), TB, 0, stream>>>(X, T1, T0, Ef);             // x += gate*res
    }

    // ---- output heads ------------------------------------------------------
    k_heads<<<grid((long)E * 4), TB, 0, stream>>>(X, P[46], P[47], P[48], P[49],
                                                  Dout, Wout);
}
